// TransformerModel_4612794876249
// MI455X (gfx1250) — matrix operand order repaired and checked by reading, not compiled, not run on hardware
//
#include <hip/hip_runtime.h>
#include <hip/hip_bf16.h>

typedef __attribute__((ext_vector_type(16))) _Float16 v16h;
typedef __attribute__((ext_vector_type(8)))  float    v8f;

constexpr int D_  = 512;
constexpr int H_  = 8;
constexpr int HD_ = 64;
constexpr int FF_ = 2048;
constexpr int V_  = 32000;
constexpr int LE_ = 2;
constexpr int LD_ = 2;
constexpr int B_  = 4;
constexpr int S_  = 1024;
constexpr int M_  = B_ * S_;   // 4096 tokens

// ---------------------------------------------------------------- WMMA helpers

__device__ __forceinline__ v8f wmma_f16(v16h a, v16h b, v8f c) {
  // D = A(16x32 f16) * B(32x16 f16) + C(16x16 f32)
  return __builtin_amdgcn_wmma_f32_16x16x32_f16(false, a, false, b, (short)0, c,
                                                false, false);
}

// A-matrix fragment, rows m = lane&15 from row-major [16+][ld] f16.
// halves 0..7 -> K = g*8 + 0..7 ; halves 8..15 -> K = 16 + g*8 + 0..7  (g = lane>>4)
__device__ __forceinline__ v16h frag_a(const _Float16* p, int ld) {
  const int lane = threadIdx.x & 31;
  const _Float16* r = p + (size_t)(lane & 15) * ld + ((lane >> 4) << 3);
  union { v16h v; uint4 q[2]; } u;
  u.q[0] = *(const uint4*)(r);
  u.q[1] = *(const uint4*)(r + 16);
  return u.v;
}

// B-matrix fragment read from [N][K] row-major: col n = lane&15,
// halves 0..7 -> K = g*8 + 0..7 ; halves 8..15 -> K = 16 + g*8 + 0..7  (g = lane>>4)
__device__ __forceinline__ v16h frag_b(const _Float16* p, int ld) {
  const int lane = threadIdx.x & 31;
  const _Float16* r = p + (size_t)(lane & 15) * ld + ((lane >> 4) << 3);
  union { v16h v; uint4 q[2]; } u;
  u.q[0] = *(const uint4*)(r);
  u.q[1] = *(const uint4*)(r + 16);
  return u.v;
}

// ---------------------------------------------------------------- async copy helpers
// gfx1250: GLOBAL_LOAD_ASYNC_TO_LDS_B128 copies 16B/lane global->LDS without
// touching VGPRs; tracked by ASYNCcnt. Generic pointers to __shared__ carry the
// LDS byte offset in their low 32 bits (aperture layout), which is what the
// instruction's LDS-address VGPR wants.

__device__ __forceinline__ unsigned lds_off(const void* p) {
  return (unsigned)(unsigned long long)(uintptr_t)p;
}

__device__ __forceinline__ void async_b128(unsigned lds, const void* gaddr) {
  asm volatile("global_load_async_to_lds_b128 %0, %1, off"
               :: "v"(lds), "v"(gaddr) : "memory");
}

// ---------------------------------------------------------------- GEMM
// C[M,N] = A_h[M,K] @ Wt_h[N,K]^T + bias ; writes f32 (Cf) and/or f16 (Ch)
// Double-buffered LDS, async global->LDS staging pipelined against WMMA.

__global__ __launch_bounds__(256) void k_gemm(const _Float16* __restrict__ A,
                                              const _Float16* __restrict__ Bt,
                                              const float* __restrict__ bias,
                                              float* __restrict__ Cf,
                                              _Float16* __restrict__ Ch,
                                              int M, int N, int K) {
  __shared__ __align__(16) _Float16 As[2][128][32];
  __shared__ __align__(16) _Float16 Bs[2][128][32];
  const int t    = threadIdx.x;
  const int bm   = blockIdx.y * 128;
  const int bn   = blockIdx.x * 128;
  const int w    = t >> 5;
  const int lane = t & 31;
  const int wm   = (w & 3) * 32;   // 4 waves along M (32 rows each)
  const int wn   = (w >> 2) * 64;  // 2 waves along N (64 cols each)

  v8f acc[2][4];
#pragma unroll
  for (int i = 0; i < 2; i++)
#pragma unroll
    for (int j = 0; j < 4; j++)
#pragma unroll
      for (int r = 0; r < 8; r++) acc[i][j][r] = 0.f;

  const int sr = t >> 1;           // 0..127 staging row
  const int sc = (t & 1) * 16;     // 0 / 16 staging col
  const _Float16* gA = A  + (size_t)(bm + sr) * K + sc;
  const _Float16* gB = Bt + (size_t)(bn + sr) * K + sc;

  // prologue: stage first K-tile into buffer 0
  async_b128(lds_off(&As[0][sr][sc]),     gA);
  async_b128(lds_off(&As[0][sr][sc + 8]), gA + 8);
  async_b128(lds_off(&Bs[0][sr][sc]),     gB);
  async_b128(lds_off(&Bs[0][sr][sc + 8]), gB + 8);

  int cur = 0;
  for (int kk = 0; kk < K; kk += 32, cur ^= 1) {
    if (kk + 32 < K) {
      // stage next K-tile into the other buffer, then wait only for current
      const int nxt = cur ^ 1;
      async_b128(lds_off(&As[nxt][sr][sc]),     gA + kk + 32);
      async_b128(lds_off(&As[nxt][sr][sc + 8]), gA + kk + 40);
      async_b128(lds_off(&Bs[nxt][sr][sc]),     gB + kk + 32);
      async_b128(lds_off(&Bs[nxt][sr][sc + 8]), gB + kk + 40);
      asm volatile("s_wait_asynccnt 0x4" ::: "memory");
    } else {
      asm volatile("s_wait_asynccnt 0x0" ::: "memory");
    }
    __syncthreads();   // publish this wave's completed async copies to the WGP

    v16h af[2], bf[4];
#pragma unroll
    for (int i = 0; i < 2; i++) af[i] = frag_a(&As[cur][wm + i * 16][0], 32);
#pragma unroll
    for (int j = 0; j < 4; j++) bf[j] = frag_b(&Bs[cur][wn + j * 16][0], 32);
#pragma unroll
    for (int i = 0; i < 2; i++)
#pragma unroll
      for (int j = 0; j < 4; j++) acc[i][j] = wmma_f16(af[i], bf[j], acc[i][j]);

    __syncthreads();   // all reads of buffer `cur` done before it is re-staged
  }

  const int lg = lane >> 4;
  const int ln = lane & 15;
#pragma unroll
  for (int j = 0; j < 4; j++) {
    const int col = bn + wn + j * 16 + ln;
    const float bv = bias[col];
#pragma unroll
    for (int i = 0; i < 2; i++)
#pragma unroll
      for (int r = 0; r < 8; r++) {
        const int row = bm + wm + i * 16 + r + 8 * lg;
        const float v = acc[i][j][r] + bv;
        if (Cf) Cf[(size_t)row * N + col] = v;
        if (Ch) Ch[(size_t)row * N + col] = (_Float16)v;
      }
  }
}

// ---------------------------------------------------------------- Flash attention
// Q,K,V f16 [B,S,H,HD]; O f16 [B,S,D]. One block = 4 waves = 64 query rows.

__global__ __launch_bounds__(128) void k_attn(const _Float16* __restrict__ Q,
                                              const _Float16* __restrict__ Kk,
                                              const _Float16* __restrict__ Vv,
                                              _Float16* __restrict__ O) {
  __shared__ __align__(16) _Float16 Vt[64][32];      // [d][key] transposed V chunk
  __shared__ __align__(16) _Float16 Pl[4][16][32];   // per-wave probability tile
  const int bh    = blockIdx.y;
  const int b     = bh >> 3;           // H_ == 8
  const int h     = bh & 7;
  const int w     = threadIdx.x >> 5;
  const int lane  = threadIdx.x & 31;
  const int lg    = lane >> 4;
  const int ln    = lane & 15;
  const int qbase = blockIdx.x * 64 + w * 16;
  const int ld    = H_ * HD_;          // 512

  const _Float16* qp = Q + ((size_t)(b * S_ + qbase)) * ld + h * HD_;
  const v16h qf0 = frag_a(qp, ld);       // d 0..31
  const v16h qf1 = frag_a(qp + 32, ld);  // d 32..63

  float mrun[8], lrun[8];
  v8f oac[4];
#pragma unroll
  for (int r = 0; r < 8; r++) { mrun[r] = -3.0e38f; lrun[r] = 0.f; }
#pragma unroll
  for (int j = 0; j < 4; j++)
#pragma unroll
    for (int r = 0; r < 8; r++) oac[j][r] = 0.f;

  const float scale = 0.125f;            // 1/sqrt(HD)
  const int vkey = threadIdx.x >> 2;     // 0..31
  const int vds  = (threadIdx.x & 3) * 16;

  for (int kb = 0; kb < S_; kb += 32) {
    { // cooperative transpose-stage of V chunk: Vt[d][key]
      const _Float16* vp = Vv + ((size_t)(b * S_ + kb + vkey)) * ld + h * HD_ + vds;
      union { uint4 q[2]; _Float16 e[16]; } u;
      u.q[0] = *(const uint4*)vp;
      u.q[1] = *(const uint4*)(vp + 8);
#pragma unroll
      for (int j = 0; j < 16; j++) Vt[vds + j][vkey] = u.e[j];
    }
    __syncthreads();

    // scores: two 16x16 key tiles, contraction over HD=64 (2 WMMAs each)
    const _Float16* kp = Kk + ((size_t)(b * S_ + kb)) * ld + h * HD_;
    v8f s[2];
#pragma unroll
    for (int kt = 0; kt < 2; kt++) {
      const v16h kf0 = frag_b(kp + (size_t)(kt * 16) * ld, ld);
      const v16h kf1 = frag_b(kp + (size_t)(kt * 16) * ld + 32, ld);
      v8f c;
#pragma unroll
      for (int r = 0; r < 8; r++) c[r] = 0.f;
      c = wmma_f16(qf0, kf0, c);
      c = wmma_f16(qf1, kf1, c);
      s[kt] = c;
    }

    // online softmax update (row stats via half-wave shuffles)
#pragma unroll
    for (int r = 0; r < 8; r++) {
      const float a0 = s[0][r] * scale;
      const float a1 = s[1][r] * scale;
      float mx = fmaxf(a0, a1);
      mx = fmaxf(mx, __shfl_xor(mx, 1, 32));
      mx = fmaxf(mx, __shfl_xor(mx, 2, 32));
      mx = fmaxf(mx, __shfl_xor(mx, 4, 32));
      mx = fmaxf(mx, __shfl_xor(mx, 8, 32));
      const float mnew  = fmaxf(mrun[r], mx);
      const float alpha = __expf(mrun[r] - mnew);
      const float p0 = __expf(a0 - mnew);
      const float p1 = __expf(a1 - mnew);
      Pl[w][r + 8 * lg][ln]      = (_Float16)p0;
      Pl[w][r + 8 * lg][16 + ln] = (_Float16)p1;
      float rs = p0 + p1;
      rs += __shfl_xor(rs, 1, 32);
      rs += __shfl_xor(rs, 2, 32);
      rs += __shfl_xor(rs, 4, 32);
      rs += __shfl_xor(rs, 8, 32);
      lrun[r] = lrun[r] * alpha + rs;
      mrun[r] = mnew;
#pragma unroll
      for (int j = 0; j < 4; j++) oac[j][r] *= alpha;
    }

    // O += P(16x32) @ V(32x64) ; P via LDS round-trip, V from transposed stage
    const v16h pf = frag_a(&Pl[w][0][0], 32);
#pragma unroll
    for (int j = 0; j < 4; j++) {
      const v16h vf = frag_b(&Vt[j * 16][0], 32);
      oac[j] = wmma_f16(pf, vf, oac[j]);
    }
    __syncthreads();   // before restaging Vt
  }

#pragma unroll
  for (int j = 0; j < 4; j++)
#pragma unroll
    for (int r = 0; r < 8; r++) {
      const int row = qbase + r + 8 * lg;
      const int col = h * HD_ + j * 16 + ln;
      O[((size_t)(b * S_ + row)) * ld + col] = (_Float16)(oac[j][r] / lrun[r]);
    }
}

// ---------------------------------------------------------------- LayerNorm(res + delta)

__global__ __launch_bounds__(256) void k_lnres(const float* __restrict__ res,
                                               const float* __restrict__ dlt,
                                               const float* __restrict__ g,
                                               const float* __restrict__ bt,
                                               float* __restrict__ of,
                                               _Float16* __restrict__ oh) {
  __shared__ float red[256];
  const int row = blockIdx.x;
  const int t   = threadIdx.x;
  const size_t base = (size_t)row * D_;
  const float v0 = res[base + t]       + dlt[base + t];
  const float v1 = res[base + 256 + t] + dlt[base + 256 + t];
  red[t] = v0 + v1;
  __syncthreads();
  for (int st = 128; st > 0; st >>= 1) { if (t < st) red[t] += red[t + st]; __syncthreads(); }
  const float mu = red[0] * (1.f / D_);
  __syncthreads();
  const float c0 = v0 - mu, c1 = v1 - mu;
  red[t] = c0 * c0 + c1 * c1;
  __syncthreads();
  for (int st = 128; st > 0; st >>= 1) { if (t < st) red[t] += red[t + st]; __syncthreads(); }
  const float inv = rsqrtf(red[0] * (1.f / D_) + 1e-5f);
  const float o0 = g[t] * c0 * inv + bt[t];
  const float o1 = g[256 + t] * c1 * inv + bt[256 + t];
  of[base + t] = o0;             of[base + 256 + t] = o1;
  oh[base + t] = (_Float16)o0;   oh[base + 256 + t] = (_Float16)o1;
}

// ---------------------------------------------------------------- embedding + PE

__global__ __launch_bounds__(256) void k_embed(const int* __restrict__ tok,
                                               const float* __restrict__ emb,
                                               float* __restrict__ of,
                                               _Float16* __restrict__ oh) {
  const int row = blockIdx.x;           // token index in [0, B*S)
  const int b   = row / S_;             // PE uses BATCH index (reference quirk)
  const int t   = threadIdx.x;
  const int id  = tok[row];
  const float sc = 22.62741699796952f;  // sqrt(512)
#pragma unroll
  for (int j = 0; j < 2; j++) {
    const int d = t + j * 256;
    const float ang = (float)b * __expf(-(logf(10000.f) / (float)D_) * (float)(d & ~1));
    const float pe  = (d & 1) ? cosf(ang) : sinf(ang);
    const float v   = emb[(size_t)id * D_ + d] * sc + pe;
    of[(size_t)row * D_ + d] = v;
    oh[(size_t)row * D_ + d] = (_Float16)v;
  }
}

// ---------------------------------------------------------------- exact GELU -> f16

__global__ __launch_bounds__(256) void k_gelu(const float* __restrict__ in,
                                              _Float16* __restrict__ out, int n) {
  const int i = blockIdx.x * 256 + threadIdx.x;
  if (i < n) {
    const float x = in[i];
    out[i] = (_Float16)(0.5f * x * (1.f + erff(x * 0.70710678118654752f)));
  }
}

// ---------------------------------------------------------------- weight fp32[K][N] -> f16[N][K]
// K is always a power of two here (512 or 2048): kshift = log2(K)

__global__ __launch_bounds__(256) void k_wconv(const float* __restrict__ w,
                                               _Float16* __restrict__ wt,
                                               int kshift, int N) {
  const size_t i = (size_t)blockIdx.x * 256 + threadIdx.x;
  const int K = 1 << kshift;
  if (i < ((size_t)N << kshift)) {
    const int n = (int)(i >> kshift);
    const int k = (int)(i & (K - 1));
    wt[i] = (_Float16)w[(size_t)k * N + n];
  }
}

// ---------------------------------------------------------------- orchestration

extern "C" void kernel_launch(void* const* d_in, const int* in_sizes, int n_in,
                              void* d_out, int out_size, void* d_ws, size_t ws_size,
                              hipStream_t stream) {
  const int*   src        = (const int*)  d_in[0];
  const int*   tgt        = (const int*)  d_in[1];
  const float* emb_enc    = (const float*)d_in[2];
  const float* emb_dec    = (const float*)d_in[3];
  const float* enc_attn_w = (const float*)d_in[4];
  const float* enc_attn_b = (const float*)d_in[5];
  const float* enc_w1     = (const float*)d_in[6];
  const float* enc_b1     = (const float*)d_in[7];
  const float* enc_w2     = (const float*)d_in[8];
  const float* enc_b2     = (const float*)d_in[9];
  const float* enc_ln_g   = (const float*)d_in[10];
  const float* enc_ln_b   = (const float*)d_in[11];
  const float* dec_attn_w = (const float*)d_in[12];
  const float* dec_attn_b = (const float*)d_in[13];
  const float* dec_w1     = (const float*)d_in[14];
  const float* dec_b1     = (const float*)d_in[15];
  const float* dec_w2     = (const float*)d_in[16];
  const float* dec_b2     = (const float*)d_in[17];
  const float* dec_ln_g   = (const float*)d_in[18];
  const float* dec_ln_b   = (const float*)d_in[19];
  const float* fc_w       = (const float*)d_in[20];
  const float* fc_b       = (const float*)d_in[21];

  size_t off = 0;
  auto alloc = [&](size_t bytes) -> void* {
    void* p = (char*)d_ws + off;
    off += (bytes + 255) & ~(size_t)255;
    return p;
  };
  _Float16* Wea = (_Float16*)alloc((size_t)LE_ * 4 * D_ * D_ * 2);
  _Float16* We1 = (_Float16*)alloc((size_t)LE_ * D_ * FF_  * 2);
  _Float16* We2 = (_Float16*)alloc((size_t)LE_ * FF_ * D_  * 2);
  _Float16* Wda = (_Float16*)alloc((size_t)LD_ * 8 * D_ * D_ * 2);
  _Float16* Wd1 = (_Float16*)alloc((size_t)LD_ * D_ * FF_  * 2);
  _Float16* Wd2 = (_Float16*)alloc((size_t)LD_ * FF_ * D_  * 2);
  _Float16* Wfc = (_Float16*)alloc((size_t)D_ * V_         * 2);
  float*    x   = (float*)   alloc((size_t)M_ * D_  * 4);
  float*    y   = (float*)   alloc((size_t)M_ * D_  * 4);
  float*    t0  = (float*)   alloc((size_t)M_ * D_  * 4);
  float*    t1  = (float*)   alloc((size_t)M_ * FF_ * 4);
  _Float16* xh  = (_Float16*)alloc((size_t)M_ * D_  * 2);
  _Float16* yh  = (_Float16*)alloc((size_t)M_ * D_  * 2);
  _Float16* qh  = (_Float16*)alloc((size_t)M_ * D_  * 2);
  _Float16* kh  = (_Float16*)alloc((size_t)M_ * D_  * 2);
  _Float16* vh  = (_Float16*)alloc((size_t)M_ * D_  * 2);
  _Float16* aoh = (_Float16*)alloc((size_t)M_ * D_  * 2);
  _Float16* fh  = (_Float16*)alloc((size_t)M_ * FF_ * 2);

  auto cgrid = [](size_t n) { return (unsigned)((n + 255) / 256); };
  constexpr int LOG_D  = 9;   // 512
  constexpr int LOG_FF = 11;  // 2048

  // --- weight convert + transpose to f16 [N][K]
  for (int i = 0; i < LE_ * 4; i++)
    k_wconv<<<cgrid((size_t)D_ * D_), 256, 0, stream>>>(
        enc_attn_w + (size_t)i * D_ * D_, Wea + (size_t)i * D_ * D_, LOG_D, D_);
  for (int i = 0; i < LE_; i++) {
    k_wconv<<<cgrid((size_t)D_ * FF_), 256, 0, stream>>>(
        enc_w1 + (size_t)i * D_ * FF_, We1 + (size_t)i * D_ * FF_, LOG_D, FF_);
    k_wconv<<<cgrid((size_t)FF_ * D_), 256, 0, stream>>>(
        enc_w2 + (size_t)i * FF_ * D_, We2 + (size_t)i * FF_ * D_, LOG_FF, D_);
  }
  for (int i = 0; i < LD_ * 8; i++)
    k_wconv<<<cgrid((size_t)D_ * D_), 256, 0, stream>>>(
        dec_attn_w + (size_t)i * D_ * D_, Wda + (size_t)i * D_ * D_, LOG_D, D_);
  for (int i = 0; i < LD_; i++) {
    k_wconv<<<cgrid((size_t)D_ * FF_), 256, 0, stream>>>(
        dec_w1 + (size_t)i * D_ * FF_, Wd1 + (size_t)i * D_ * FF_, LOG_D, FF_);
    k_wconv<<<cgrid((size_t)FF_ * D_), 256, 0, stream>>>(
        dec_w2 + (size_t)i * FF_ * D_, Wd2 + (size_t)i * FF_ * D_, LOG_FF, D_);
  }
  k_wconv<<<cgrid((size_t)D_ * V_), 256, 0, stream>>>(fc_w, Wfc, LOG_D, V_);

  // --- embeddings
  k_embed<<<M_, 256, 0, stream>>>(src, emb_enc, x, xh);
  k_embed<<<M_, 256, 0, stream>>>(tgt, emb_dec, y, yh);

  const dim3 gD(D_ / 128, M_ / 128);
  const dim3 gF(FF_ / 128, M_ / 128);
  const dim3 gV(V_ / 128, M_ / 128);
  const dim3 gA(S_ / 64, B_ * H_);

  // --- encoder
  for (int i = 0; i < LE_; i++) {
    const _Float16* aw = Wea + (size_t)i * 4 * D_ * D_;
    const float*    ab = enc_attn_b + (size_t)i * 4 * D_;
    k_gemm<<<gD, 256, 0, stream>>>(xh, aw,                       ab,          nullptr, qh, M_, D_, D_);
    k_gemm<<<gD, 256, 0, stream>>>(xh, aw + (size_t)1 * D_ * D_, ab + D_,     nullptr, kh, M_, D_, D_);
    k_gemm<<<gD, 256, 0, stream>>>(xh, aw + (size_t)2 * D_ * D_, ab + 2 * D_, nullptr, vh, M_, D_, D_);
    k_attn<<<gA, 128, 0, stream>>>(qh, kh, vh, aoh);
    k_gemm<<<gD, 256, 0, stream>>>(aoh, aw + (size_t)3 * D_ * D_, ab + 3 * D_, t0, nullptr, M_, D_, D_);
    k_lnres<<<M_, 256, 0, stream>>>(x, t0, enc_ln_g + (size_t)i * 2 * D_,
                                    enc_ln_b + (size_t)i * 2 * D_, x, xh);
    k_gemm<<<gF, 256, 0, stream>>>(xh, We1 + (size_t)i * D_ * FF_,
                                   enc_b1 + (size_t)i * FF_, t1, nullptr, M_, FF_, D_);
    k_gelu<<<cgrid((size_t)M_ * FF_), 256, 0, stream>>>(t1, fh, M_ * FF_);
    k_gemm<<<gD, 256, 0, stream>>>(fh, We2 + (size_t)i * FF_ * D_,
                                   enc_b2 + (size_t)i * D_, t0, nullptr, M_, D_, FF_);
    k_lnres<<<M_, 256, 0, stream>>>(x, t0, enc_ln_g + (size_t)i * 2 * D_ + D_,
                                    enc_ln_b + (size_t)i * 2 * D_ + D_, x, xh);
  }

  // --- decoder
  for (int i = 0; i < LD_; i++) {
    const _Float16* aw = Wda + (size_t)i * 8 * D_ * D_;
    const float*    ab = dec_attn_b + (size_t)i * 8 * D_;
    const float*    lg = dec_ln_g + (size_t)i * 3 * D_;
    const float*    lb = dec_ln_b + (size_t)i * 3 * D_;
    // self-attention (no mask in reference)
    k_gemm<<<gD, 256, 0, stream>>>(yh, aw,                       ab,          nullptr, qh, M_, D_, D_);
    k_gemm<<<gD, 256, 0, stream>>>(yh, aw + (size_t)1 * D_ * D_, ab + D_,     nullptr, kh, M_, D_, D_);
    k_gemm<<<gD, 256, 0, stream>>>(yh, aw + (size_t)2 * D_ * D_, ab + 2 * D_, nullptr, vh, M_, D_, D_);
    k_attn<<<gA, 128, 0, stream>>>(qh, kh, vh, aoh);
    k_gemm<<<gD, 256, 0, stream>>>(aoh, aw + (size_t)3 * D_ * D_, ab + 3 * D_, t0, nullptr, M_, D_, D_);
    k_lnres<<<M_, 256, 0, stream>>>(y, t0, lg, lb, y, yh);
    // cross-attention (K/V from encoder output x)
    k_gemm<<<gD, 256, 0, stream>>>(yh, aw + (size_t)4 * D_ * D_, ab + 4 * D_, nullptr, qh, M_, D_, D_);
    k_gemm<<<gD, 256, 0, stream>>>(xh, aw + (size_t)5 * D_ * D_, ab + 5 * D_, nullptr, kh, M_, D_, D_);
    k_gemm<<<gD, 256, 0, stream>>>(xh, aw + (size_t)6 * D_ * D_, ab + 6 * D_, nullptr, vh, M_, D_, D_);
    k_attn<<<gA, 128, 0, stream>>>(qh, kh, vh, aoh);
    k_gemm<<<gD, 256, 0, stream>>>(aoh, aw + (size_t)7 * D_ * D_, ab + 7 * D_, t0, nullptr, M_, D_, D_);
    k_lnres<<<M_, 256, 0, stream>>>(y, t0, lg + D_, lb + D_, y, yh);
    // FFN
    k_gemm<<<gF, 256, 0, stream>>>(yh, Wd1 + (size_t)i * D_ * FF_,
                                   dec_b1 + (size_t)i * FF_, t1, nullptr, M_, FF_, D_);
    k_gelu<<<cgrid((size_t)M_ * FF_), 256, 0, stream>>>(t1, fh, M_ * FF_);
    k_gemm<<<gD, 256, 0, stream>>>(fh, Wd2 + (size_t)i * FF_ * D_,
                                   dec_b2 + (size_t)i * D_, t0, nullptr, M_, D_, FF_);
    k_lnres<<<M_, 256, 0, stream>>>(y, t0, lg + 2 * D_, lb + 2 * D_, y, yh);
  }

  // --- final projection to vocab (dominant GEMM, 134 GF)
  k_gemm<<<gV, 256, 0, stream>>>(yh, Wfc, fc_b, (float*)d_out, nullptr, M_, V_, D_);

  (void)in_sizes; (void)n_in; (void)out_size; (void)ws_size;
}
